// AcousticTextEncoderLayer_78108275245158
// MI455X (gfx1250) — matrix operand order repaired and checked by reading, not compiled, not run on hardware
//
#include <hip/hip_runtime.h>
#include <hip/hip_bf16.h>
#include <math.h>

// Problem constants (match reference)
#define BB 16
#define TT 4096
#define HH 512
#define KK 9
#define TP (TT + 8)          // padded time (4-row zero halo each side)
#define LN_EPS 1e-5f
#define SLOPE 0.1f

typedef __attribute__((ext_vector_type(16))) __bf16 v16bf;
typedef __attribute__((ext_vector_type(8)))  __bf16 v8bf;
typedef __attribute__((ext_vector_type(8)))  float  v8f;

typedef unsigned int u32x4 __attribute__((ext_vector_type(4)));
typedef int          i32x4 __attribute__((ext_vector_type(4)));
typedef int          i32x8 __attribute__((ext_vector_type(8)));

// --- TDM availability probe (compile-time safe) -----------------------------
#if defined(__has_builtin)
#if __has_builtin(__builtin_amdgcn_tensor_load_to_lds) && \
    __has_builtin(__builtin_amdgcn_s_wait_tensorcnt)
#define USE_TDM 1
#endif
#endif
#ifndef USE_TDM
#define USE_TDM 0
#endif

// ---------------------------------------------------------------------------
// Kernel 1: weight norm  w = g * v / ||v||  (norm over H_in*K per out channel)
// and repack fp32 [co][ci][k] -> bf16 Wt[k][co][ci] (ci contiguous = WMMA K dim)
// ---------------------------------------------------------------------------
__global__ __launch_bounds__(256) void prep_weight(
    const float* __restrict__ wv, const float* __restrict__ wg,
    __bf16* __restrict__ Wt)
{
    const int co  = blockIdx.x;       // 0..511
    const int tid = threadIdx.x;      // 0..255
    const size_t base = (size_t)co * (HH * KK);   // 4608 elems per out channel

    __shared__ float red[256];
    __shared__ float sc;

    float s = 0.f;
    for (int idx = tid; idx < HH * KK; idx += 256) {
        float v = wv[base + idx];
        s += v * v;
    }
    red[tid] = s;
    __syncthreads();
    for (int off = 128; off > 0; off >>= 1) {
        if (tid < off) red[tid] += red[tid + off];
        __syncthreads();
    }
    if (tid == 0) sc = wg[co] / sqrtf(red[0]);
    __syncthreads();
    const float scale = sc;

    for (int idx = tid; idx < HH * KK; idx += 256) {
        int ci = idx / KK;
        int k  = idx - ci * KK;
        float w = wv[base + idx] * scale;
        Wt[((size_t)(k * HH + co)) * HH + ci] = (__bf16)w;
    }
}

// ---------------------------------------------------------------------------
// Kernel 2: masked input -> bf16 with zero time-halo: Xb[b][t+4][c]
// ---------------------------------------------------------------------------
__global__ __launch_bounds__(256) void prep_input(
    const float* __restrict__ hs, const int* __restrict__ lens,
    __bf16* __restrict__ Xb)
{
    size_t idx = (size_t)blockIdx.x * 256 + threadIdx.x;
    const size_t total = (size_t)BB * TP * HH;
    if (idx >= total) return;
    int c    = (int)(idx & (HH - 1));
    size_t r = idx >> 9;              // / HH
    int tp   = (int)(r % TP);
    int b    = (int)(r / TP);
    int t    = tp - 4;
    float v = 0.f;
    if (t >= 0 && t < TT && t < lens[b])
        v = hs[((size_t)b * TT + t) * HH + c];
    Xb[idx] = (__bf16)v;
}

// ---------------------------------------------------------------------------
// Kernel 3: conv-as-GEMM with bf16 WMMA, fp32 accumulation.
// Block = 256 threads = 8 waves; block tile = 128 (time) x 64 (c_out).
// Waves stacked in M: wave w computes rows [w*16, w*16+16) x all 64 c_out
// (4 accumulators -> 4 independent WMMA chains, A fragment reused 4x).
// K-loop: 16 chunks of 32 input channels x 9 taps.
// ---------------------------------------------------------------------------
#define A_ROWS   136         // 128 + 8 halo rows
#define A_STRIDE 40          // padded rows: 32 data + 8 pad bf16 (16B pad)
#define B_STRIDE 40

#if USE_TDM
// Issue one 2D TDM tile load: tile_rows x 32 bf16 elements from a row-major
// bf16 tensor (row pitch = pitch_elems), into LDS at lds_off with 16B row
// padding (32 elems -> 40-elem LDS stride, matching A_STRIDE/B_STRIDE).
__device__ __forceinline__ void tdm_load_tile(
    const __bf16* gsrc, unsigned lds_off, unsigned tile_rows,
    unsigned tensor_rows, unsigned pitch_elems)
{
    unsigned long long ga = (unsigned long long)(uintptr_t)gsrc;
    u32x4 g0;
    g0.x = 1u;                                        // count=1 descriptor
    g0.y = lds_off;                                   // LDS byte address
    g0.z = (unsigned)(ga & 0xFFFFFFFFull);            // global_addr[31:0]
    g0.w = (unsigned)((ga >> 32) & 0x1FFFFFFull)      // global_addr[56:32]
         | (2u << 30);                                // type = 2 ("image")

    // group1: data_size=1 (2B), pad_enable, pad_interval=3 (16 DW = 64B row),
    // pad_amount=3 (4 DW = 16B)  => LDS row stride 80B = 40 bf16
    unsigned w0 = (1u << 16) | (1u << 20) | (3u << 22) | (3u << 25);
    unsigned w1 = (pitch_elems & 0xFFFFu) << 16;              // tensor_dim0 lo
    unsigned w2 = ((pitch_elems >> 16) & 0xFFFFu)             // tensor_dim0 hi
                | ((tensor_rows & 0xFFFFu) << 16);            // tensor_dim1 lo
    unsigned w3 = ((tensor_rows >> 16) & 0xFFFFu)             // tensor_dim1 hi
                | (32u << 16);                                // tile_dim0 = 32
    unsigned w4 = (tile_rows & 0xFFFFu);                      // tile_dim1
    unsigned w5 = pitch_elems;                                // dim0_stride lo32
    i32x8 g1 = { (int)w0, (int)w1, (int)w2, (int)w3,
                 (int)w4, (int)w5, 0, 0 };
    i32x4 gz = { 0, 0, 0, 0 };
#if defined(__clang_major__) && (__clang_major__ >= 23)
    i32x8 gz8 = { 0, 0, 0, 0, 0, 0, 0, 0 };
    __builtin_amdgcn_tensor_load_to_lds(g0, g1, gz, gz, gz8, 0);
#else
    __builtin_amdgcn_tensor_load_to_lds(g0, g1, gz, gz, 0);
#endif
}
#endif

__global__ __launch_bounds__(256) void conv_gemm(
    const __bf16* __restrict__ Xb, const __bf16* __restrict__ Wt,
    const float* __restrict__ bias, float* __restrict__ Y)
{
    __shared__ __align__(16) __bf16 Asmem[A_ROWS * A_STRIDE];
    __shared__ __align__(16) __bf16 Bsmem[KK * 64 * B_STRIDE];

    const int tid   = threadIdx.x;
    const int t0    = blockIdx.x * 128;
    const int n_blk = blockIdx.y * 64;
    const int b     = blockIdx.z;

    const int wave = tid >> 5, lane = tid & 31;
    const int m0w  = wave * 16;             // wave's time sub-tile (M)
    const int half = lane >> 4;             // K-group selector per layout
    const int l15  = lane & 15;

    v8f acc0 = {}, acc1 = {}, acc2 = {}, acc3 = {};

    // Xb time index t+4 == conv input row t; tap k reads rows t0 .. t0+135.
    const __bf16* Xbase = Xb + ((size_t)b * TP + t0) * HH;

    // defeat "shared array never written" analysis when TDM writes via DMA
    if (bias == nullptr) { Asmem[0] = (__bf16)0.f; Bsmem[0] = (__bf16)0.f; }

    for (int cc = 0; cc < 16; ++cc) {
        const int c0 = cc * 32;
        __syncthreads();   // previous chunk's LDS reads complete

#if USE_TDM
        // Wave 0 issues the 10 tile DMAs (TDM ignores EXEC; one issue/wave).
        if (tid < 32) {
            tdm_load_tile(&Xbase[c0], (unsigned)(uintptr_t)Asmem,
                          A_ROWS, TP, HH);
            #pragma unroll
            for (int k = 0; k < KK; ++k) {
                tdm_load_tile(&Wt[((size_t)k * HH + n_blk) * HH + c0],
                              (unsigned)(uintptr_t)&Bsmem[k * 64 * B_STRIDE],
                              64, HH, HH);
            }
        }
        __builtin_amdgcn_s_wait_tensorcnt(0);   // issuing wave drains TENSORcnt
        __syncthreads();                        // tiles visible to all waves
#else
        // Stage A: 136 rows x 32 ch, 8-elem (16B) segments: 544 tasks
        for (int task = tid; task < A_ROWS * 4; task += 256) {
            int row = task >> 2, seg = task & 3;
            *(v8bf*)&Asmem[row * A_STRIDE + seg * 8] =
                *(const v8bf*)&Xbase[(size_t)row * HH + c0 + seg * 8];
        }
        // Stage B: 9 taps x (64 rows x 32 ch) = 2304 segments (9 per thread)
        for (int task = tid; task < 2304; task += 256) {
            int k   = task >> 8;
            int rem = task & 255;
            int row = rem >> 2, seg = rem & 3;
            *(v8bf*)&Bsmem[(k * 64 + row) * B_STRIDE + seg * 8] =
                *(const v8bf*)&Wt[((size_t)k * HH + n_blk + row) * HH + c0 + seg * 8];
        }
        // Prefetch next chunk's A region (global_prefetch_b8)
        if (cc < 15 && tid < A_ROWS)
            __builtin_prefetch(&Xbase[(size_t)tid * HH + c0 + 32], 0, 0);
        __syncthreads();   // tiles visible
#endif

        #pragma unroll
        for (int k = 0; k < KK; ++k) {
            // A fragment (16x32 bf16): lanes 0-15 -> K {0..7,16..23},
            // lanes 16-31 -> K {8..15,24..31}; row = m + k (halo-shifted).
            const __bf16* ap = &Asmem[(m0w + l15 + k) * A_STRIDE + half * 8];
            v8bf alo = *(const v8bf*)ap;          // K = half*8 + 0..7
            v8bf ahi = *(const v8bf*)(ap + 16);   // K = half*8 + 16..23
            v16bf a;
            #pragma unroll
            for (int i = 0; i < 8; ++i) { a[i] = alo[i]; a[8 + i] = ahi[i]; }

            // 4 B fragments (32x16 bf16): lane = N column,
            // lanes 0-15 hold K {0..7,16..23}, lanes 16-31 hold K {8..15,24..31}.
            const __bf16* brow = &Bsmem[(k * 64 + l15) * B_STRIDE + half * 8];
            v16bf bf[4];
            #pragma unroll
            for (int s = 0; s < 4; ++s) {
                const __bf16* bp = brow + s * 16 * B_STRIDE;
                v8bf blo = *(const v8bf*)bp;
                v8bf bhi = *(const v8bf*)(bp + 16);
                #pragma unroll
                for (int i = 0; i < 8; ++i) { bf[s][i] = blo[i]; bf[s][8 + i] = bhi[i]; }
            }

            acc0 = __builtin_amdgcn_wmma_f32_16x16x32_bf16(
                false, a, false, bf[0], (short)0, acc0, false, false);
            acc1 = __builtin_amdgcn_wmma_f32_16x16x32_bf16(
                false, a, false, bf[1], (short)0, acc1, false, false);
            acc2 = __builtin_amdgcn_wmma_f32_16x16x32_bf16(
                false, a, false, bf[2], (short)0, acc2, false, false);
            acc3 = __builtin_amdgcn_wmma_f32_16x16x32_bf16(
                false, a, false, bf[3], (short)0, acc3, false, false);
        }
    }

    // Epilogue: C/D layout — VGPR r: lanes 0-15 M=r, lanes 16-31 M=r+8.
    const int n0 = n_blk + l15;
    const float bia0 = bias[n0];
    const float bia1 = bias[n0 + 16];
    const float bia2 = bias[n0 + 32];
    const float bia3 = bias[n0 + 48];
    #pragma unroll
    for (int r = 0; r < 8; ++r) {
        int m = t0 + m0w + r + half * 8;
        size_t base = ((size_t)b * TT + m) * HH;
        Y[base + n0]      = acc0[r] + bia0;
        Y[base + n0 + 16] = acc1[r] + bia1;
        Y[base + n0 + 32] = acc2[r] + bia2;
        Y[base + n0 + 48] = acc3[r] + bia3;
    }
}

// ---------------------------------------------------------------------------
// Kernel 4: mask-select (conv vs raw input), LayerNorm over H, leaky ReLU.
// One block per (b,t) row; in-place on d_out (read row -> reduce -> write).
// ---------------------------------------------------------------------------
__global__ __launch_bounds__(256) void mask_ln_act(
    const float* __restrict__ hs, const int* __restrict__ lens,
    const float* __restrict__ gamma, const float* __restrict__ beta,
    float* y)
{
    const int row = blockIdx.x;          // 0 .. B*T-1
    const int b   = row >> 12;           // / 4096
    const int t   = row & (TT - 1);
    const int tid = threadIdx.x;
    const size_t base = (size_t)row * HH;

    const bool valid = t < lens[b];
    float v0, v1;
    if (valid) { v0 = y[base + tid];  v1 = y[base + tid + 256]; }
    else       { v0 = hs[base + tid]; v1 = hs[base + tid + 256]; }

    __shared__ float s1[256];
    __shared__ float s2[256];
    s1[tid] = v0 + v1;
    s2[tid] = v0 * v0 + v1 * v1;
    __syncthreads();
    for (int off = 128; off > 0; off >>= 1) {
        if (tid < off) { s1[tid] += s1[tid + off]; s2[tid] += s2[tid + off]; }
        __syncthreads();
    }
    const float mean = s1[0] * (1.0f / HH);
    const float var  = s2[0] * (1.0f / HH) - mean * mean;
    const float rstd = rsqrtf(var + LN_EPS);

    float o0 = (v0 - mean) * rstd * gamma[tid]       + beta[tid];
    float o1 = (v1 - mean) * rstd * gamma[tid + 256] + beta[tid + 256];
    o0 = (o0 >= 0.f) ? o0 : SLOPE * o0;
    o1 = (o1 >= 0.f) ? o1 : SLOPE * o1;
    y[base + tid]       = o0;
    y[base + tid + 256] = o1;
}

// ---------------------------------------------------------------------------
extern "C" void kernel_launch(void* const* d_in, const int* in_sizes, int n_in,
                              void* d_out, int out_size, void* d_ws, size_t ws_size,
                              hipStream_t stream)
{
    const float* hs    = (const float*)d_in[0];  // [B,T,H]
    const int*   lens  = (const int*)  d_in[1];  // [B]
    const float* wv    = (const float*)d_in[2];  // [H,H,K]
    const float* wg    = (const float*)d_in[3];  // [H,1,1]
    const float* cb    = (const float*)d_in[4];  // [H]
    const float* gamma = (const float*)d_in[5];  // [H]
    const float* beta  = (const float*)d_in[6];  // [H]
    float* y = (float*)d_out;                    // [B,T,H]

    // Workspace layout: Xb bf16 [B][T+8][H] (67.2 MB) + Wt bf16 [K][H][H] (4.7 MB)
    __bf16* Xb = (__bf16*)d_ws;
    __bf16* Wt = Xb + (size_t)BB * TP * HH;

    prep_weight<<<HH, 256, 0, stream>>>(wv, wg, Wt);

    const size_t n_in_elems = (size_t)BB * TP * HH;
    prep_input<<<(unsigned)((n_in_elems + 255) / 256), 256, 0, stream>>>(hs, lens, Xb);

    dim3 g3(TT / 128, HH / 64, BB);   // 32 x 8 x 16 = 4096 blocks
    conv_gemm<<<g3, 256, 0, stream>>>(Xb, Wt, cb, y);

    mask_ln_act<<<BB * TT, 256, 0, stream>>>(hs, lens, gamma, beta, y);
}
